// YOLOv1Acc_16930761080834
// MI455X (gfx1250) — matrix operand order repaired and checked by reading, not compiled, not run on hardware
//
#include <hip/hip_runtime.h>
#include <hip/hip_bf16.h>
#include <stdint.h>

typedef __attribute__((ext_vector_type(16))) _Float16 v16h;
typedef __attribute__((ext_vector_type(8)))  float    v8f;

#define NBOX 49      // S*S boxes per image
#define NC   20      // classes
#define NPI  1470    // floats per image (7*7*30)

__device__ __forceinline__ float sigf(float x) { return 1.0f / (1.0f + expf(-x)); }

__device__ __forceinline__ float iou_xyxy(const float* a, const float* b) {
    float lx = fmaxf(a[0], b[0]);
    float ly = fmaxf(a[1], b[1]);
    float rx = fminf(a[2], b[2]);
    float ry = fminf(a[3], b[3]);
    float w  = fmaxf(rx - lx, 0.0f);
    float h  = fmaxf(ry - ly, 0.0f);
    float inter = w * h;
    float aa = (a[2] - a[0]) * (a[3] - a[1]);
    float ab = (b[2] - b[0]) * (b[3] - b[1]);
    return inter / (aa + ab - inter);
}

// ---------------------------------------------------------------------------
// Kernel A: one workgroup (64 thr = 2 waves) per image.
// Decodes, sorts, NMS, matches; emits per-entry class/tp/vp and per-class
// counts (valid-pred, tp, gt) via one-hot-histogram WMMA matmuls.
// ---------------------------------------------------------------------------
__global__ __launch_bounds__(64) void yolo_per_image(
    const float* __restrict__ tgt, const float* __restrict__ outp, int batch,
    uint32_t* __restrict__ cls_g,
    unsigned long long* __restrict__ tpb_g,
    unsigned long long* __restrict__ vpb_g,
    float* __restrict__ cnt_m, float* __restrict__ cnt_t, float* __restrict__ cnt_g)
{
    const int img = blockIdx.x;
    const int tid = threadIdx.x;

    __shared__ float so[NPI];
    __shared__ float st[NPI];
    __shared__ float ob[NBOX][4], tb[NBOX][4];
    __shared__ float oc[NBOX], tconf[NBOX];
    __shared__ int   ocls[NBOX], tcls[NBOX];
    __shared__ float bs[NBOX][4], cs[NBOX];
    __shared__ int   ks[NBOX], order[NBOX];
    __shared__ unsigned long long suprow[NBOX];
    __shared__ float mx[NBOX];
    __shared__ int   best[NBOX];
    __shared__ unsigned long long bitres[2];   // [0]=valid_p bits, [1]=tp bits

    const float* tg = tgt + (size_t)img * NPI;
    const float* og = outp + (size_t)img * NPI;

    if (img + 1 < batch) {                     // stream-ahead: global_prefetch_b8
        __builtin_prefetch(tg + NPI + tid * 32, 0, 0);
        __builtin_prefetch(og + NPI + tid * 32, 0, 0);
    }
    for (int i = tid; i < NPI; i += 64) { so[i] = og[i]; st[i] = tg[i]; }
    __syncthreads();

    // ---- per-cell decode ----
    if (tid < NBOX) {
        const int n  = tid;
        const int iy = n / 7, jx = n % 7;
        const float* t = &st[n * 30];
        const float* o = &so[n * 30];
        const float gw = 448.0f / 7.0f;

        // ground truth
        tconf[n] = t[4];
        { int ac = 0; float av = t[10];
          for (int c = 1; c < NC; c++) { float v = t[10 + c]; if (v > av) { av = v; ac = c; } }
          tcls[n] = ac; }
        { float x = t[0], y = t[1], w = t[2] * 448.0f, h = t[3] * 448.0f;
          float cx = (x + (float)jx) * gw, cy = (y + (float)iy) * gw;
          tb[n][0] = cx - 0.5f * w; tb[n][1] = cy - 0.5f * h;
          tb[n][2] = cx + 0.5f * w; tb[n][3] = cy + 0.5f * h; }

        // prediction (responsible box = max-conf anchor; ties -> anchor 0)
        float c0 = sigf(o[4]), c1 = sigf(o[9]);
        int r = (c0 >= c1) ? 0 : 1;
        oc[n] = fmaxf(c0, c1);
        { float x = sigf(o[r * 5 + 0]), y = sigf(o[r * 5 + 1]);
          float w = sigf(o[r * 5 + 2]) * 448.0f, h = sigf(o[r * 5 + 3]) * 448.0f;
          float cx = (x + (float)jx) * gw, cy = (y + (float)iy) * gw;
          ob[n][0] = cx - 0.5f * w; ob[n][1] = cy - 0.5f * h;
          ob[n][2] = cx + 0.5f * w; ob[n][3] = cy + 0.5f * h; }
        { int ac = 0; float av = o[10];   // sigmoid is monotonic -> raw argmax
          for (int c = 1; c < NC; c++) { float v = o[10 + c]; if (v > av) { av = v; ac = c; } }
          ocls[n] = ac; }
    }
    __syncthreads();

    // ---- stable sort by conf desc (rank scatter) ----
    if (tid < NBOX) {
        float ci = oc[tid]; int rank = 0;
        for (int j = 0; j < NBOX; j++) {
            float cj = oc[j];
            if (cj > ci || (cj == ci && j < tid)) rank++;
        }
        order[rank] = tid;
    }
    __syncthreads();
    if (tid < NBOX) {
        int s = order[tid];
        bs[tid][0] = ob[s][0]; bs[tid][1] = ob[s][1];
        bs[tid][2] = ob[s][2]; bs[tid][3] = ob[s][3];
        cs[tid] = oc[s]; ks[tid] = ocls[s];
    }
    __syncthreads();

    // ---- NMS suppression rows + best same-class GT IoU ----
    if (tid < NBOX) {
        unsigned long long m = 0;
        for (int j = 0; j < NBOX; j++) {
            float v = iou_xyxy(bs[tid], bs[j]);
            if (v > 0.5f && ks[tid] == ks[j]) m |= 1ull << j;
        }
        suprow[tid] = m;

        float mb = -1.0f; int bi = 0;
        for (int j = 0; j < NBOX; j++) {
            if (ks[tid] == tcls[j] && tconf[j] > 0.5f) {
                float v = iou_xyxy(bs[tid], tb[j]);
                if (v > mb) { mb = v; bi = j; }
            }
        }
        mx[tid] = mb; best[tid] = bi;
    }
    __syncthreads();

    // ---- serial NMS scan + greedy matching (49 steps, one lane) ----
    if (tid == 0) {
        unsigned long long suppressed = 0, keep = 0;
        for (int i = 0; i < NBOX; i++) {
            if (!((suppressed >> i) & 1ull)) {
                keep |= 1ull << i;
                suppressed |= suprow[i] & (~0ull << (i + 1));
            }
        }
        unsigned long long vp = 0;
        for (int i = 0; i < NBOX; i++)
            if (((keep >> i) & 1ull) && cs[i] > 0.5f) vp |= 1ull << i;
        unsigned long long matched = 0, tp = 0;
        for (int i = 0; i < NBOX; i++) {
            if (((vp >> i) & 1ull) && mx[i] > 0.5f &&
                !((matched >> best[i]) & 1ull)) {
                matched |= 1ull << best[i];
                tp |= 1ull << i;
            }
        }
        bitres[0] = vp; bitres[1] = tp;
    }
    __syncthreads();

    if (tid < NBOX) cls_g[(size_t)img * NBOX + tid] = (uint32_t)ks[tid];
    if (tid == 0) { vpb_g[img] = bitres[0]; tpb_g[img] = bitres[1]; }

    // ---- WMMA one-hot histogram: D[q][c] = sum_k A[q][k] * onehot(cls_k==c)
    // K axis hosts 98 entries: k<49 = sorted preds, 49..97 = GT cells.
    // Rows: q0=valid_p, q1=tp, q2=valid_g. K padded to 128 = 4 chunks of 32.
    if (tid < 32) {                            // wave 0, EXEC all ones
        const unsigned long long vp = bitres[0], tp = bitres[1];
        const int half = tid >> 4;             // lane half selects K stripes
        const int m    = tid & 15;             // A row / B column (class mod 16)

        union HV { v16h v; _Float16 h[16]; };
        HV A[4];
        #pragma unroll
        for (int cc = 0; cc < 4; cc++) {
            #pragma unroll
            for (int e = 0; e < 16; e++) {
                // 16-bit A 16x32 layout: lanes<16 hold K={0..7,16..23}, lanes>=16 K={8..15,24..31}
                int kk = cc * 32 + (half == 0 ? (e < 8 ? e : e + 8)
                                              : (e < 8 ? e + 8 : e + 16));
                float val = 0.0f;
                if (m == 0)      { if (kk < NBOX && ((vp >> kk) & 1ull)) val = 1.0f; }
                else if (m == 1) { if (kk < NBOX && ((tp >> kk) & 1ull)) val = 1.0f; }
                else if (m == 2) { if (kk >= NBOX && kk < 2 * NBOX &&
                                       tconf[kk - NBOX] > 0.5f)          val = 1.0f; }
                A[cc].h[e] = (_Float16)val;
            }
        }
        #pragma unroll
        for (int tt = 0; tt < 2; tt++) {       // class tiles 0..15, 16..31
            v8f acc = {};
            const int cb = m + tt * 16;
            #pragma unroll
            for (int cc = 0; cc < 4; cc++) {
                HV B;
                #pragma unroll
                for (int e = 0; e < 16; e++) {
                    // 16-bit B 32x16 layout: lanes<16 hold K={0..7,16..23}, lanes>=16 K={8..15,24..31}
                    int kk = cc * 32 + (half == 0 ? (e < 8 ? e : e + 8)
                                                  : (e < 8 ? e + 8 : e + 16));
                    int kc = -1;
                    if (kk < NBOX)          kc = ks[kk];
                    else if (kk < 2 * NBOX) kc = tcls[kk - NBOX];
                    B.h[e] = (_Float16)((kc == cb) ? 1.0f : 0.0f);
                }
                acc = __builtin_amdgcn_wmma_f32_16x16x32_f16(
                        false, A[cc].v, false, B.v, (short)0, acc, false, false);
            }
            // f32 C/D layout: lanes 0-15 -> N=lane, acc[v] = row M=v
            if (half == 0 && cb < NC) {
                cnt_m[(size_t)img * NC + cb] = acc[0];
                cnt_t[(size_t)img * NC + cb] = acc[1];
                cnt_g[(size_t)img * NC + cb] = acc[2];
            }
        }
    }
}

// ---------------------------------------------------------------------------
// Kernel B: per-class exclusive scan over images of (mask,tp) counts + gt sum.
// One block per class, 256 threads x 16 images each (batch <= 4096).
// ---------------------------------------------------------------------------
__global__ __launch_bounds__(256) void class_scan(
    const float* __restrict__ cnt_m, const float* __restrict__ cnt_t,
    const float* __restrict__ cnt_g, int batch,
    float* __restrict__ excl_m, float* __restrict__ excl_t,
    float* __restrict__ gt_count)
{
    const int c = blockIdx.x;
    const int t = threadIdx.x;
    const int PER = 16;

    float lm[PER], lt[PER];
    float sm = 0.0f, stp = 0.0f, sg = 0.0f;
    #pragma unroll
    for (int i = 0; i < PER; i++) {
        int img = t * PER + i;
        float vm = 0.0f, vt = 0.0f, vg = 0.0f;
        if (img < batch) {
            vm = cnt_m[(size_t)img * NC + c];
            vt = cnt_t[(size_t)img * NC + c];
            vg = cnt_g[(size_t)img * NC + c];
        }
        lm[i] = sm; lt[i] = stp;
        sm += vm; stp += vt; sg += vg;
    }
    __shared__ float shm[256], sht[256], shg[256];
    shm[t] = sm; sht[t] = stp; shg[t] = sg;
    __syncthreads();
    for (int off = 1; off < 256; off <<= 1) {
        float vm = (t >= off) ? shm[t - off] : 0.0f;
        float vt = (t >= off) ? sht[t - off] : 0.0f;
        __syncthreads();
        shm[t] += vm; sht[t] += vt;
        __syncthreads();
    }
    float exm = (t == 0) ? 0.0f : shm[t - 1];
    float ext = (t == 0) ? 0.0f : sht[t - 1];
    #pragma unroll
    for (int i = 0; i < PER; i++) {
        int img = t * PER + i;
        if (img < batch) {
            excl_m[(size_t)img * NC + c] = exm + lm[i];
            excl_t[(size_t)img * NC + c] = ext + lt[i];
        }
    }
    for (int off = 128; off > 0; off >>= 1) {
        __syncthreads();
        if (t < off) shg[t] += shg[t + off];
    }
    if (t == 0) gt_count[c] = shg[0];
}

// ---------------------------------------------------------------------------
// Kernel C: per image, per class: closed-form trapz terms at TP entries.
// AP_c * (gt+1e-6) = sum over TPs of 0.5*(p_k + p_{k-1}).
// ---------------------------------------------------------------------------
__global__ __launch_bounds__(32) void ap_terms(
    const uint32_t* __restrict__ cls_g,
    const unsigned long long* __restrict__ tpb_g,
    const unsigned long long* __restrict__ vpb_g,
    const float* __restrict__ excl_m, const float* __restrict__ excl_t,
    float* __restrict__ partial)
{
    const int img = blockIdx.x;
    const int c   = threadIdx.x;
    if (c >= NC) return;

    const unsigned long long vp = vpb_g[img];
    const unsigned long long tp = tpb_g[img];
    const float baseK = excl_m[(size_t)img * NC + c];
    const float baseT = excl_t[(size_t)img * NC + c];

    int lk = 0, lt = 0;
    float acc = 0.0f;
    for (int i = 0; i < NBOX; i++) {
        if (((vp >> i) & 1ull) && cls_g[(size_t)img * NBOX + i] == (uint32_t)c) {
            lk++;
            if ((tp >> i) & 1ull) {
                lt++;
                float K = baseK + (float)lk;     // masked entries so far (incl.)
                float T = baseT + (float)lt;     // TP entries so far (incl.)
                float p = T / (K + 1e-6f);
                // prepended p0=1 applies only to the global first flat entry
                float pprev = (img == 0 && i == 0)
                                  ? 1.0f
                                  : (T - 1.0f) / ((K - 1.0f) + 1e-6f);
                acc += 0.5f * (p + pprev);
            }
        }
    }
    partial[(size_t)img * NC + c] = acc;
}

// ---------------------------------------------------------------------------
// Kernel D: deterministic per-class reduction of partial terms.
// ---------------------------------------------------------------------------
__global__ __launch_bounds__(256) void class_reduce(
    const float* __restrict__ partial, int batch, float* __restrict__ apsum)
{
    const int c = blockIdx.x;
    const int t = threadIdx.x;
    float s = 0.0f;
    for (int img = t; img < batch; img += 256)
        s += partial[(size_t)img * NC + c];
    __shared__ float sh[256];
    sh[t] = s;
    for (int off = 128; off > 0; off >>= 1) {
        __syncthreads();
        if (t < off) sh[t] += sh[t + off];
    }
    if (t == 0) apsum[c] = sh[0];
}

// ---------------------------------------------------------------------------
// Kernel E: mAP = mean over classes with gt_count>0 of apsum/(gt+1e-6).
// ---------------------------------------------------------------------------
__global__ void finalize(const float* __restrict__ apsum,
                         const float* __restrict__ gt_count,
                         float* __restrict__ out)
{
    if (threadIdx.x != 0 || blockIdx.x != 0) return;
    float s = 0.0f, n = 0.0f;
    for (int c = 0; c < NC; c++) {
        float g = gt_count[c];
        if (g > 0.0f) { s += apsum[c] / (g + 1e-6f); n += 1.0f; }
    }
    out[0] = s / fmaxf(n, 1.0f);
}

// ---------------------------------------------------------------------------
extern "C" void kernel_launch(void* const* d_in, const int* in_sizes, int n_in,
                              void* d_out, int out_size, void* d_ws, size_t ws_size,
                              hipStream_t stream) {
    (void)n_in; (void)out_size; (void)ws_size;
    const float* target = (const float*)d_in[0];
    const float* output = (const float*)d_in[1];
    float* out = (float*)d_out;
    const int batch = in_sizes[0] / NPI;     // 4096

    char* p = (char*)d_ws;
    auto alloc = [&](size_t bytes) -> char* {
        char* r = p;
        p += (bytes + 255) & ~(size_t)255;
        return r;
    };
    uint32_t*           cls_g  = (uint32_t*)          alloc((size_t)batch * NBOX * sizeof(uint32_t));
    unsigned long long* tpb    = (unsigned long long*)alloc((size_t)batch * sizeof(unsigned long long));
    unsigned long long* vpb    = (unsigned long long*)alloc((size_t)batch * sizeof(unsigned long long));
    float*              cnt_m  = (float*)             alloc((size_t)batch * NC * sizeof(float));
    float*              cnt_t  = (float*)             alloc((size_t)batch * NC * sizeof(float));
    float*              cnt_g  = (float*)             alloc((size_t)batch * NC * sizeof(float));
    float*              excl_m = (float*)             alloc((size_t)batch * NC * sizeof(float));
    float*              excl_t = (float*)             alloc((size_t)batch * NC * sizeof(float));
    float*              gt_cnt = (float*)             alloc(NC * sizeof(float));
    float*              part   = (float*)             alloc((size_t)batch * NC * sizeof(float));
    float*              apsum  = (float*)             alloc(NC * sizeof(float));

    hipLaunchKernelGGL(yolo_per_image, dim3(batch), dim3(64), 0, stream,
                       target, output, batch, cls_g, tpb, vpb, cnt_m, cnt_t, cnt_g);
    hipLaunchKernelGGL(class_scan, dim3(NC), dim3(256), 0, stream,
                       cnt_m, cnt_t, cnt_g, batch, excl_m, excl_t, gt_cnt);
    hipLaunchKernelGGL(ap_terms, dim3(batch), dim3(32), 0, stream,
                       cls_g, tpb, vpb, excl_m, excl_t, part);
    hipLaunchKernelGGL(class_reduce, dim3(NC), dim3(256), 0, stream,
                       part, batch, apsum);
    hipLaunchKernelGGL(finalize, dim3(1), dim3(1), 0, stream,
                       apsum, gt_cnt, out);
}
